// DGCL_14207751815717
// MI455X (gfx1250) — compile-verified, matrix operand order checked by reading, not run on hardware
//
#include <hip/hip_runtime.h>

// DGCL forward for MI455X (gfx1250, wave32, WMMA).
// Collapses the 8 disentangled heads (K*d = 128 = F) into two extra 128-wide
// GIN layers with block-diagonal weights, so all 5 layers run the same fused
// WMMA f16 GEMM kernel. Edge aggregation (the bandwidth-dominant part,
// ~4 GB of gather+atomic traffic) uses float4 gathers with 32 lanes/edge and
// global_atomic_add_f32 scatter.

#define NN 50000
#define EE 800000
#define FF 128
#define GG 512
#define NLAYER 5
#define BN_EPS 1e-5f

typedef __attribute__((ext_vector_type(16))) _Float16 v16h;
typedef __attribute__((ext_vector_type(8)))  float    v8f;

__global__ void zero_f32(float* p, int n) {
  int i = blockIdx.x * blockDim.x + threadIdx.x;
  for (; i < n; i += gridDim.x * blockDim.x) p[i] = 0.f;
}

// Build f16, transposed ([out_col][in_col]) weights for all 5 layers.
// Layers 0..2: gc_W1/gc_W2 dense.  Layer 3: W1 = concat_k h0_W1[k] (dense),
// W2 = blockdiag(h0_W2).  Layer 4: blockdiag(h1_W1), blockdiag(h1_W2).
__global__ void prep_weights(const float* __restrict__ gcW1, const float* __restrict__ gcW2,
                             const float* __restrict__ h0W1, const float* __restrict__ h0W2,
                             const float* __restrict__ h1W1, const float* __restrict__ h1W2,
                             _Float16* __restrict__ Wt) {
  int t = blockIdx.x * blockDim.x + threadIdx.x;   // layout: [l][m][c][f]
  if (t >= NLAYER * 2 * FF * FF) return;
  int f = t & 127;              // input feature
  int c = (t >> 7) & 127;       // output feature (row of transposed weight)
  int m = (t >> 14) & 1;        // 0 = W1, 1 = W2
  int l = t >> 15;              // layer
  float v = 0.f;
  if (l < 3) {
    const float* W = m ? gcW2 : gcW1;          // [GC][F][F]
    v = W[l * FF * FF + f * FF + c];           // transpose
  } else {
    int k = c >> 4, j = c & 15;
    if (l == 3 && m == 0) {
      v = h0W1[k * FF * 16 + f * 16 + j];      // [K][F][d], dense concat
    } else {
      const float* W = (l == 3) ? h0W2 : (m == 0 ? h1W1 : h1W2);  // [K][d][d]
      if ((f >> 4) == k) v = W[k * 256 + (f & 15) * 16 + j];      // block-diag
    }
  }
  Wt[t] = (_Float16)v;
}

// agg[dst] += feat[src] over all edges; 32 lanes per edge, float4 per lane.
__global__ void edge_agg(const float* __restrict__ feat, const int* __restrict__ src,
                         const int* __restrict__ dst, float* __restrict__ agg) {
  int gid = blockIdx.x * blockDim.x + threadIdx.x;
  int e = gid >> 5;
  if (e >= EE) return;
  int part = (gid & 31) << 2;
  int s = src[e], d = dst[e];
  const float4 v = *(const float4*)(feat + (size_t)s * FF + part);
  float* ap = agg + (size_t)d * FF + part;
  atomicAdd(ap + 0, v.x);
  atomicAdd(ap + 1, v.y);
  atomicAdd(ap + 2, v.z);
  atomicAdd(ap + 3, v.w);
}

// Fused GIN MLP:  out = relu((hin+agg) @ W1t^T + b1) @ W2t^T + b2
// One wave -> one 16-row x 128-col tile.  GEMM1 output goes through a
// per-wave LDS region (f16) to re-layout D-frags into A-frags for GEMM2.
__global__ void __launch_bounds__(256) gin_mlp(
    const float* __restrict__ hin, const float* __restrict__ agg,
    const _Float16* __restrict__ W1t, const _Float16* __restrict__ W2t,
    const float* __restrict__ b1, const float* __restrict__ b2,
    float* __restrict__ out) {
  __shared__ __align__(32) _Float16 lds[8][16][FF];   // 32 KB, private per wave
  const int lane  = threadIdx.x & 31;
  const int wid   = threadIdx.x >> 5;
  const int tile  = blockIdx.x * 8 + wid;
  if (tile * 16 >= NN) return;                        // no block-wide sync used
  const int mrow  = lane & 15;        // A row within tile / B-D column index
  const int khalf = (lane >> 4) << 4; // 0 for lanes 0-15, 16 for lanes 16-31
  const int rhalf = khalf >> 1;       // 0 or 8: D-matrix row offset
  const int row0  = tile * 16;

  // A fragments for GEMM1: z = hin + agg, fp32 -> f16, K = 4 chunks of 32
  v16h afrag[4];
#pragma unroll
  for (int ch = 0; ch < 4; ++ch) {
    const size_t base = (size_t)(row0 + mrow) * FF + ch * 32 + khalf;
#pragma unroll
    for (int q = 0; q < 4; ++q) {
      float4 hv = *(const float4*)(hin + base + q * 4);
      float4 av = *(const float4*)(agg + base + q * 4);
      afrag[ch][q * 4 + 0] = (_Float16)(hv.x + av.x);
      afrag[ch][q * 4 + 1] = (_Float16)(hv.y + av.y);
      afrag[ch][q * 4 + 2] = (_Float16)(hv.z + av.z);
      afrag[ch][q * 4 + 3] = (_Float16)(hv.w + av.w);
    }
  }

  // GEMM1 (+bias via C-init, ReLU) -> LDS f16
#pragma unroll
  for (int ct = 0; ct < 8; ++ct) {
    const int col = ct * 16 + mrow;
    const float bv = b1[col];
    v8f acc;
#pragma unroll
    for (int r = 0; r < 8; ++r) acc[r] = bv;
#pragma unroll
    for (int ch = 0; ch < 4; ++ch) {
      v16h bfrag = *(const v16h*)(W1t + (size_t)col * FF + ch * 32 + khalf);
      acc = __builtin_amdgcn_wmma_f32_16x16x32_f16(false, afrag[ch], false, bfrag,
                                                   (short)0, acc, false, false);
    }
#pragma unroll
    for (int r = 0; r < 8; ++r) {
      float y = acc[r] > 0.f ? acc[r] : 0.f;
      lds[wid][r + rhalf][col] = (_Float16)y;
    }
  }

  // same-wave cross-lane RAW through LDS: drain DS stores before the reloads
  asm volatile("s_wait_dscnt 0x0" ::: "memory");

  // A fragments for GEMM2 from LDS
  v16h a2[4];
#pragma unroll
  for (int ch = 0; ch < 4; ++ch)
    a2[ch] = *(const v16h*)&lds[wid][mrow][ch * 32 + khalf];

  // GEMM2 (+bias), fp32 store
#pragma unroll
  for (int ct = 0; ct < 8; ++ct) {
    const int col = ct * 16 + mrow;
    const float bv = b2[col];
    v8f acc;
#pragma unroll
    for (int r = 0; r < 8; ++r) acc[r] = bv;
#pragma unroll
    for (int ch = 0; ch < 4; ++ch) {
      v16h bfrag = *(const v16h*)(W2t + (size_t)col * FF + ch * 32 + khalf);
      acc = __builtin_amdgcn_wmma_f32_16x16x32_f16(false, a2[ch], false, bfrag,
                                                   (short)0, acc, false, false);
    }
#pragma unroll
    for (int r = 0; r < 8; ++r)
      out[(size_t)(row0 + r + rhalf) * FF + col] = acc[r];
  }
}

// Column-wise sum and sum-of-squares over the node axis (training-mode BN).
__global__ void bn_stats(const float* __restrict__ hraw, float* __restrict__ stats) {
  int t = blockIdx.x * blockDim.x + threadIdx.x;
  int c = t & 127;
  int r0 = t >> 7;
  const int rstride = (gridDim.x * blockDim.x) >> 7;
  float s = 0.f, ss = 0.f;
  for (int r = r0; r < NN; r += rstride) {
    float v = hraw[(size_t)r * FF + c];
    s += v;
    ss += v * v;
  }
  atomicAdd(&stats[c], s);
  atomicAdd(&stats[FF + c], ss);
}

__global__ void bn_apply(const float* __restrict__ hraw, const float* __restrict__ stats,
                         const float* __restrict__ gam, const float* __restrict__ bet,
                         int do_relu, float* __restrict__ hout) {
  int i = blockIdx.x * blockDim.x + threadIdx.x;
  if (i >= NN * FF) return;
  int c = i & 127;
  const float inv_n = 1.f / (float)NN;
  float mu  = stats[c] * inv_n;
  float var = stats[FF + c] * inv_n - mu * mu;
  float y = (hraw[i] - mu) * rsqrtf(var + BN_EPS) * gam[c] + bet[c];
  if (do_relu) y = fmaxf(y, 0.f);
  hout[i] = y;
}

// global_add_pool: out[batch[n]][c] += h[n][c]; col c = k*16+j matches [G,K,d].
__global__ void pool_kernel(const float* __restrict__ h, const int* __restrict__ batch,
                            float* __restrict__ out) {
  int i = blockIdx.x * blockDim.x + threadIdx.x;
  if (i >= NN * FF) return;
  int n = i >> 7, c = i & 127;
  atomicAdd(&out[(size_t)batch[n] * FF + c], h[i]);
}

extern "C" void kernel_launch(void* const* d_in, const int* in_sizes, int n_in,
                              void* d_out, int out_size, void* d_ws, size_t ws_size,
                              hipStream_t stream) {
  const float* x     = (const float*)d_in[0];
  const int*   ei    = (const int*)d_in[1];
  const int*   batch = (const int*)d_in[2];
  const float* gcW1 = (const float*)d_in[4];
  const float* gcb1 = (const float*)d_in[5];
  const float* gcW2 = (const float*)d_in[6];
  const float* gcb2 = (const float*)d_in[7];
  const float* gcg  = (const float*)d_in[8];
  const float* gcbe = (const float*)d_in[9];
  const float* h0W1 = (const float*)d_in[10];
  const float* h0b1 = (const float*)d_in[11];
  const float* h0W2 = (const float*)d_in[12];
  const float* h0b2 = (const float*)d_in[13];
  const float* h0g  = (const float*)d_in[14];
  const float* h0be = (const float*)d_in[15];
  const float* h1W1 = (const float*)d_in[16];
  const float* h1b1 = (const float*)d_in[17];
  const float* h1W2 = (const float*)d_in[18];
  const float* h1b2 = (const float*)d_in[19];
  const float* h1g  = (const float*)d_in[20];
  const float* h1be = (const float*)d_in[21];

  // workspace carve-out (~77 MB total)
  char* ws = (char*)d_ws;
  size_t off = 0;
  auto carve = [&](size_t bytes) -> void* {
    void* p = ws + off;
    off = (off + bytes + 255) & ~(size_t)255;
    return p;
  };
  float*    hA    = (float*)carve((size_t)NN * FF * 4);   // BN output / layer input
  float*    aggb  = (float*)carve((size_t)NN * FF * 4);   // edge aggregation
  float*    hraw  = (float*)carve((size_t)NN * FF * 4);   // pre-BN GEMM output
  float*    stats = (float*)carve(2 * FF * 4);            // sum | sumsq
  _Float16* Wt    = (_Float16*)carve((size_t)NLAYER * 2 * FF * FF * 2);

  const int* src = ei;
  const int* dst = ei + EE;

  prep_weights<<<(NLAYER * 2 * FF * FF + 255) / 256, 256, 0, stream>>>(
      gcW1, gcW2, h0W1, h0W2, h1W1, h1W2, Wt);

  const float* b1s[NLAYER] = {gcb1, gcb1 + FF, gcb1 + 2 * FF, h0b1, h1b1};
  const float* b2s[NLAYER] = {gcb2, gcb2 + FF, gcb2 + 2 * FF, h0b2, h1b2};
  const float* gs [NLAYER] = {gcg,  gcg  + FF, gcg  + 2 * FF, h0g,  h1g };
  const float* bes[NLAYER] = {gcbe, gcbe + FF, gcbe + 2 * FF, h0be, h1be};
  const int    relus[NLAYER] = {1, 1, 0, 1, 0};

  const int nElem = NN * FF;
  for (int l = 0; l < NLAYER; ++l) {
    const float* cur = (l == 0) ? x : hA;
    zero_f32<<<1024, 256, 0, stream>>>(aggb, nElem);
    edge_agg<<<(EE * 32) / 256, 256, 0, stream>>>(cur, src, dst, aggb);
    gin_mlp<<<(NN / 16 + 7) / 8, 256, 0, stream>>>(
        cur, aggb,
        Wt + (size_t)(l * 2 + 0) * FF * FF,
        Wt + (size_t)(l * 2 + 1) * FF * FF,
        b1s[l], b2s[l], hraw);
    zero_f32<<<1, 256, 0, stream>>>(stats, 2 * FF);
    bn_stats<<<200, 256, 0, stream>>>(hraw, stats);
    bn_apply<<<(nElem + 255) / 256, 256, 0, stream>>>(hraw, stats, gs[l], bes[l],
                                                      relus[l], hA);
  }

  zero_f32<<<64, 256, 0, stream>>>((float*)d_out, GG * FF);
  pool_kernel<<<(nElem + 255) / 256, 256, 0, stream>>>(hA, batch, (float*)d_out);
}
